// NeighborhoodModel_50276887167644
// MI455X (gfx1250) — matrix operand order repaired and checked by reading, not compiled, not run on hardware
//
#include <hip/hip_runtime.h>
#include <hip/hip_bf16.h>
#include <math.h>

// ============================================================================
// NeighborhoodModel forward for MI455X (gfx1250, wave32, WMMA).
//
// Shapes: B=8, N=1024, K=2 hops, D=64, H=4, LAYERS=2, FLOW/DUAL iters = 10.
//
// Design (see analysis):
//  * Dominant GEMM agg[k,b] = nbr[k,b](1024x1024) @ t[b](1024x64) runs on
//    v_wmma_f32_16x16x32_f16 (nbr is exactly {0,1}; t=tanh in [-1,1]; f32 acc).
//  * sparsemax reduced to per-row threshold tau via wave-parallel bisection;
//    prop_ij = adj_ij * relu(nw_j - tau_i) recomputed on the fly (never stored).
//  * min-cost flow fixed point reduced to per-node scale s_i; 10 masked
//    mat-vec passes over adj (L2-resident on the 192MB L2).
//  * dual flow: 10-step scalar recurrence per edge, in registers, one pass.
//  * All reductions deterministic (fixed-order trees, per-tile partials, no
//    float atomics) -> bitwise-stable output across replays.
//
// Params dict leaves assumed in JAX tree-flatten (sorted-key) order:
//  Uh,Ur,Uz,Wh,Wn,Wo,Wr,Wz,attW,attv,bh,bn,bo,br,bz,dW0,dW1,db0,db1,
//  eW0,eW1,eb0,eb1,gW0,gW1,gb0,gb1  at d_in[6..32].
//
// Workspace requirement: ~9.7 MB.
// ============================================================================

#define NB 8
#define NN 1024
#define ND 64
#define BIGV 1.0e7f

typedef __attribute__((ext_vector_type(16))) _Float16 v16h;
typedef __attribute__((ext_vector_type(8)))  _Float16 v8h;
typedef __attribute__((ext_vector_type(8)))  float    v8fl;
typedef __attribute__((ext_vector_type(8)))  float    v8f;

__device__ __forceinline__ float wave_sum32(float v) {
  #pragma unroll
  for (int o = 16; o >= 1; o >>= 1) v += __shfl_xor(v, o, 32);
  return v;
}
__device__ __forceinline__ float wave_max32(float v) {
  #pragma unroll
  for (int o = 16; o >= 1; o >>= 1) v = fmaxf(v, __shfl_xor(v, o, 32));
  return v;
}

// ---------------------------------------------------------------- encoder MLP
__global__ __launch_bounds__(64)
void k_encoder(const float* __restrict__ emb, const float* __restrict__ feat,
               const float* __restrict__ eW0, const float* __restrict__ eb0,
               const float* __restrict__ eW1, const float* __restrict__ eb1,
               float* __restrict__ enc) {
  int node = blockIdx.x;      // b*N + n
  int d = threadIdx.x;
  __shared__ float x[36];
  __shared__ float h0[64];
  if (d < 32)      x[d] = emb[(size_t)node * 32 + d];
  else if (d < 36) x[d] = feat[(size_t)node * 4 + (d - 32)];
  __syncthreads();
  float acc = eb0[d];
  #pragma unroll
  for (int c = 0; c < 36; ++c) acc += x[c] * eW0[c * 64 + d];
  h0[d] = tanhf(acc);
  __syncthreads();
  float a2 = eb1[d];
  #pragma unroll 8
  for (int c = 0; c < 64; ++c) a2 += h0[c] * eW1[c * 64 + d];
  enc[(size_t)node * 64 + d] = tanhf(a2);
}

// ------------------------------------------------- t = tanh(enc@Wn+bn), packed
// Pack directly into the WMMA B-matrix (32x16 f16) lane/VGPR layout:
// element (k_local, col): lane = 16*((k_local/8)%2) + col, e = k_local%8 + 8*(k_local/16), vgpr = e/2,
// half = e%2. One contiguous 32B load per B-tile per lane in k_agg.
__global__ __launch_bounds__(64)
void k_prep_t(const float* __restrict__ enc, const float* __restrict__ Wn,
              const float* __restrict__ bn, _Float16* __restrict__ tp) {
  int node = blockIdx.x;
  int d = threadIdx.x;
  int b = node >> 10, j = node & 1023;
  __shared__ float xr[64];
  xr[d] = enc[(size_t)node * 64 + d];
  __syncthreads();
  float acc = bn[d];
  #pragma unroll 8
  for (int c = 0; c < 64; ++c) acc += xr[c] * Wn[c * 64 + d];
  float t = tanhf(acc);
  int kc = j >> 5, r = j & 31, kg = (r >> 3) & 1, rr = (r & 7) | ((r >> 4) << 3), v = rr >> 1, h = rr & 1;
  int lane = (kg << 4) | (d & 15), dt = d >> 4;
  size_t idx = (((((size_t)b * 32 + kc) * 4 + dt) * 32 + lane) * 8 + v) * 2 + h;
  tp[idx] = (_Float16)t;
}

// ---------------------------------------------------- WMMA aggregation GEMM
// agg[k,b,i,d] = sum_j nbr[k,b,i,j] * t[b,j,d].  4 waves/WG; wave owns a
// 16x64 output strip = 4 f32 16x16 accumulator tiles; K-loop 32 chunks of 32.
__global__ __launch_bounds__(128)
void k_agg(const float* __restrict__ nbr, const _Float16* __restrict__ tp,
           float* __restrict__ agg) {
  const int lane = threadIdx.x & 31;
  const int wv   = threadIdx.x >> 5;
  const int b    = blockIdx.y;
  const int kz   = blockIdx.z;
  const int m0   = blockIdx.x * 64 + wv * 16;
  const int hi   = lane >> 4;     // A half-lane group (k%16 >= 8)
  const int mr   = lane & 15;     // A row within tile / C column
  const size_t rowbase = (((size_t)kz * 8 + b) * 1024 + (size_t)(m0 + mr)) * 1024;
  v8f acc[4];
  #pragma unroll
  for (int dt = 0; dt < 4; ++dt) acc[dt] = (v8f){0.f,0.f,0.f,0.f,0.f,0.f,0.f,0.f};

  for (int kc = 0; kc < 32; ++kc) {
    // A tile: lane holds k = kc*32 + 8*hi + [0..8) and kc*32 + 16 + 8*hi + [0..8)
    const v8fl* ap0 = (const v8fl*)(nbr + rowbase + kc * 32 + hi * 8);
    const v8fl* ap1 = (const v8fl*)(nbr + rowbase + kc * 32 + 16 + hi * 8);
    v8h h0 = __builtin_convertvector(*ap0, v8h);
    v8h h1 = __builtin_convertvector(*ap1, v8h);
    v16h a = __builtin_shufflevector(h0, h1, 0,1,2,3,4,5,6,7,8,9,10,11,12,13,14,15);
    // B tiles: contiguous 16 halfs per lane from the packed buffer
    const _Float16* tb = tp + (((size_t)b * 32 + kc) * 128 + lane) * 16;
    #pragma unroll
    for (int dt = 0; dt < 4; ++dt) {
      v16h bm = *(const v16h*)(tb + (size_t)dt * 512);
      acc[dt] = __builtin_amdgcn_wmma_f32_16x16x32_f16(
          false, a, false, bm, (short)0, acc[dt], false, false);
    }
  }
  // C/D layout: vgpr r -> row m0 + r + 8*hi, col dt*16 + mr
  const size_t obase = ((size_t)kz * 8 + b) * 1024;
  #pragma unroll
  for (int r = 0; r < 8; ++r) {
    float* op = agg + (obase + (size_t)(m0 + r + hi * 8)) * 64 + mr;
    op[0]  = acc[0][r];
    op[16] = acc[1][r];
    op[32] = acc[2][r];
    op[48] = acc[3][r];
  }
}

// ----------------------------------------------------------- hop attention
__global__ __launch_bounds__(64)
void k_attn(const float* __restrict__ agg, const float* __restrict__ attW,
            const float* __restrict__ attv, const float* __restrict__ Wo,
            const float* __restrict__ bo, float* __restrict__ nxt) {
  int node = blockIdx.x;
  int e = threadIdx.x;
  __shared__ float a0[64], a1[64], red[8][64], wk[2], mixed[64];
  a0[e] = agg[(size_t)node * 64 + e];
  a1[e] = agg[((size_t)8192 + node) * 64 + e];
  __syncthreads();
  #pragma unroll
  for (int h = 0; h < 4; ++h) {
    const float* W = attW + h * 4096;
    float p0 = 0.f, p1 = 0.f;
    for (int d = 0; d < 64; ++d) {
      float w = W[d * 64 + e];
      p0 += a0[d] * w;
      p1 += a1[d] * w;
    }
    float av = attv[h * 64 + e];
    red[h * 2 + 0][e] = tanhf(p0) * av;
    red[h * 2 + 1][e] = tanhf(p1) * av;
  }
  __syncthreads();
  for (int st = 32; st >= 1; st >>= 1) {
    if (e < st) {
      #pragma unroll
      for (int hk = 0; hk < 8; ++hk) red[hk][e] += red[hk][e + st];
    }
    __syncthreads();
  }
  if (e == 0) {   // softmax over hop axis (K=2) per head, then head-average
    float w0 = 0.f, w1 = 0.f;
    #pragma unroll
    for (int h = 0; h < 4; ++h) {
      float s0 = red[h * 2][0], s1 = red[h * 2 + 1][0];
      float m = fmaxf(s0, s1);
      float e0 = expf(s0 - m), e1 = expf(s1 - m);
      float inv = 1.f / (e0 + e1);
      w0 += e0 * inv;
      w1 += e1 * inv;
    }
    wk[0] = w0 * 0.25f;
    wk[1] = w1 * 0.25f;
  }
  __syncthreads();
  mixed[e] = wk[0] * a0[e] + wk[1] * a1[e];
  __syncthreads();
  float acc = bo[e];
  #pragma unroll 8
  for (int c = 0; c < 64; ++c) acc += mixed[c] * Wo[c * 64 + e];
  nxt[(size_t)node * 64 + e] = tanhf(acc);
}

// ------------------------------------------------------------------- GRU cell
__global__ __launch_bounds__(64)
void k_gru(const float* __restrict__ nxt, float* __restrict__ enc,
           const float* __restrict__ Wz, const float* __restrict__ Uz, const float* __restrict__ bz,
           const float* __restrict__ Wr, const float* __restrict__ Ur, const float* __restrict__ br,
           const float* __restrict__ Wh, const float* __restrict__ Uh, const float* __restrict__ bh) {
  int node = blockIdx.x;
  int d = threadIdx.x;
  __shared__ float x[64], s[64], rs[64];
  x[d] = nxt[(size_t)node * 64 + d];
  s[d] = enc[(size_t)node * 64 + d];
  __syncthreads();
  float az = bz[d], ar = br[d];
  for (int c = 0; c < 64; ++c) {
    az += x[c] * Wz[c * 64 + d] + s[c] * Uz[c * 64 + d];
    ar += x[c] * Wr[c * 64 + d] + s[c] * Ur[c * 64 + d];
  }
  float zg = 1.f / (1.f + expf(-az));
  float rg = 1.f / (1.f + expf(-ar));
  rs[d] = rg * s[d];
  __syncthreads();
  float ah = bh[d];
  for (int c = 0; c < 64; ++c) ah += x[c] * Wh[c * 64 + d] + rs[c] * Uh[c * 64 + d];
  float h = tanhf(ah);
  enc[(size_t)node * 64 + d] = zg * s[d] + (1.f - zg) * h;
}

// -------------------------------------- node_w (d-head) and dual_vars (g-head)
__global__ __launch_bounds__(64)
void k_heads(const float* __restrict__ enc,
             const float* __restrict__ dW0, const float* __restrict__ db0,
             const float* __restrict__ dW1, const float* __restrict__ db1,
             const float* __restrict__ gW0, const float* __restrict__ gb0,
             const float* __restrict__ gW1, const float* __restrict__ gb1,
             float* __restrict__ nw, float* __restrict__ gv) {
  int node = blockIdx.x;
  int tid = threadIdx.x;
  __shared__ float x[64], red[64];
  x[tid] = enc[(size_t)node * 64 + tid];
  __syncthreads();
  int t = tid & 31;
  const float* W0 = (tid < 32) ? dW0 : gW0;  // uniform per wave (wave32)
  const float* b0 = (tid < 32) ? db0 : gb0;
  const float* W1 = (tid < 32) ? dW1 : gW1;
  float acc = b0[t];
  #pragma unroll 8
  for (int c = 0; c < 64; ++c) acc += x[c] * W0[c * 32 + t];
  red[tid] = tanhf(acc) * W1[t];
  __syncthreads();
  for (int st = 16; st >= 1; st >>= 1) {
    if (t < st) red[tid] += red[tid + st];
    __syncthreads();
  }
  if (tid == 0)  nw[node] = red[0]  + db1[0];
  if (tid == 32) gv[node] = red[32] + gb1[0];
}

// ----------------------------- sparsemax row threshold via bisection (1 wave)
// z_j = adj_ij ? nw_j : -BIG.  tau solves sum_j relu(z_j - tau) = 1.
__global__ __launch_bounds__(256)
void k_sparsemax(const float* __restrict__ adj, const float* __restrict__ nw,
                 float* __restrict__ tau) {
  int wv = threadIdx.x >> 5, lane = threadIdx.x & 31;
  int row = blockIdx.x * 8 + wv;           // b*N + i
  int b = row >> 10;
  const float* ar  = adj + (size_t)row * 1024;
  const float* nwb = nw + b * 1024;
  float z[32];
  float mz = -3.0e38f;
  #pragma unroll
  for (int q = 0; q < 32; ++q) {
    int j = q * 32 + lane;
    float a = ar[j];
    float zz = (a != 0.f) ? nwb[j] : -BIGV;
    z[q] = zz;
    mz = fmaxf(mz, zz);
  }
  mz = wave_max32(mz);
  float lo = mz - 1.0f, hi = mz;           // sum(lo) >= 1 >= sum(hi)
  for (int it = 0; it < 48; ++it) {
    float mid = 0.5f * (lo + hi);
    float s = 0.f;
    #pragma unroll
    for (int q = 0; q < 32; ++q) s += fmaxf(z[q] - mid, 0.f);
    s = wave_sum32(s);
    if (s > 1.0f) lo = mid; else hi = mid;
  }
  if (lane == 0) tau[row] = 0.5f * (lo + hi);
}

__global__ __launch_bounds__(256)
void k_zero(float* __restrict__ p, int n) {
  int i = blockIdx.x * 256 + threadIdx.x;
  if (i < n) p[i] = 0.f;
}

// ------------------- one min-cost-flow iteration: s_out = relu(P^T s - demand)
// inflow_dest = sum_src adj[src,dest] * relu(nw_dest - tau_src) * s_src.
// 32 consecutive destinations per block (coalesced adj reads), 8 waves split
// the src range, fixed-order cross-wave combine -> deterministic.
__global__ __launch_bounds__(256)
void k_flow(const float* __restrict__ adj, const float* __restrict__ nw,
            const float* __restrict__ tau, const float* __restrict__ dem,
            const float* __restrict__ s_in, float* __restrict__ s_out) {
  int lane = threadIdx.x & 31, wv = threadIdx.x >> 5;
  int b = blockIdx.y;
  int dest = blockIdx.x * 32 + lane;
  float nwd = nw[b * 1024 + dest];
  float acc = 0.f;
  int j0 = wv * 128;
  for (int j = j0; j < j0 + 128; ++j) {
    float sj = s_in[b * 1024 + j];
    if (sj > 0.f) {
      float f = fmaxf(nwd - tau[b * 1024 + j], 0.f) * sj;
      acc += adj[((size_t)b * 1024 + j) * 1024 + dest] * f;
    }
  }
  __shared__ float part[8][32];
  part[wv][lane] = acc;
  __syncthreads();
  if (threadIdx.x < 32) {
    float inflow = 0.f;
    #pragma unroll
    for (int w = 0; w < 8; ++w) inflow += part[w][lane];
    s_out[b * 1024 + dest] = fmaxf(inflow - dem[b * 1024 + dest], 0.f);
  }
}

// ------------------ flow cost per 32x32 tile: sum (f_ij - adj*min(f_ij,f_ji))^2
__global__ __launch_bounds__(256)
void k_flowcost(const float* __restrict__ adj, const float* __restrict__ nw,
                const float* __restrict__ tau, const float* __restrict__ s,
                const int* __restrict__ cf, float* __restrict__ flowP) {
  int b = blockIdx.z;
  int i0 = blockIdx.x * 32, j0 = blockIdx.y * 32;
  int tx = threadIdx.x, ty = threadIdx.y;
  __shared__ float At[32][33];
  __shared__ float nwI[32], nwJ[32], tauI[32], tauJ[32], sI[32], sJ[32];
  __shared__ float red[256];
  for (int jj = ty; jj < 32; jj += 8)
    At[jj][tx] = adj[((size_t)b * 1024 + j0 + jj) * 1024 + i0 + tx];
  if (ty == 0) {
    nwI[tx]  = nw[b * 1024 + i0 + tx];
    nwJ[tx]  = nw[b * 1024 + j0 + tx];
    tauI[tx] = tau[b * 1024 + i0 + tx];
    tauJ[tx] = tau[b * 1024 + j0 + tx];
    sI[tx]   = s[b * 1024 + i0 + tx];
    sJ[tx]   = s[b * 1024 + j0 + tx];
  }
  __syncthreads();
  int corr = cf[0];
  float lsum = 0.f;
  for (int r = ty; r < 32; r += 8) {
    float a_ij = adj[((size_t)b * 1024 + i0 + r) * 1024 + j0 + tx];
    float f_ij = a_ij * fmaxf(nwJ[tx] - tauI[r], 0.f) * sI[r];
    float fc;
    if (corr) {
      float f_ji = At[tx][r] * fmaxf(nwI[r] - tauJ[tx], 0.f) * sJ[tx];
      fc = f_ij - a_ij * fminf(f_ij, f_ji);
    } else {
      fc = f_ij;
    }
    lsum += fc * fc;
  }
  int idx = ty * 32 + tx;
  red[idx] = lsum;
  __syncthreads();
  for (int st = 128; st >= 1; st >>= 1) {
    if (idx < st) red[idx] += red[idx + st];
    __syncthreads();
  }
  if (idx == 0) flowP[b * 1024 + blockIdx.x * 32 + blockIdx.y] = red[0];
}

// ------- dual flow: per-edge 10-step momentum PGD in registers + cost partial
__global__ __launch_bounds__(256)
void k_dualcost(const float* __restrict__ adj, const float* __restrict__ gv,
                float* __restrict__ dualP) {
  int b = blockIdx.z;
  int i0 = blockIdx.x * 32, j0 = blockIdx.y * 32;
  int tx = threadIdx.x, ty = threadIdx.y;
  __shared__ float gI[32], gJ[32];
  __shared__ float red[256];
  if (ty == 0) {
    gI[tx] = gv[b * 1024 + i0 + tx];
    gJ[tx] = gv[b * 1024 + j0 + tx];
  }
  __syncthreads();
  float lsum = 0.f;
  for (int r = ty; r < 32; r += 8) {
    float a = adj[((size_t)b * 1024 + i0 + r) * 1024 + j0 + tx];
    if (a != 0.f) {
      float dd = a * (gI[r] - gJ[tx]);
      float f = 0.f, acc = 0.f;
      #pragma unroll
      for (int t = 0; t < 10; ++t) {
        float grad = 2.f * f - dd;
        acc = 0.9f * acc + 0.1f * grad;
        f = fmaxf(f - acc, 0.f);
      }
      lsum += f * f - dd * f;
    }
  }
  int idx = ty * 32 + tx;
  red[idx] = lsum;
  __syncthreads();
  for (int st = 128; st >= 1; st >>= 1) {
    if (idx < st) red[idx] += red[idx + st];
    __syncthreads();
  }
  if (idx == 0) dualP[b * 1024 + blockIdx.x * 32 + blockIdx.y] = red[0];
}

// -------- final: out[b] = sum(flowP) - sum(dualP) + sum(g*demand), fixed order
__global__ __launch_bounds__(256)
void k_final(const float* __restrict__ flowP, const float* __restrict__ dualP,
             const float* __restrict__ gv, const float* __restrict__ dem,
             float* __restrict__ out) {
  int b = blockIdx.x;
  int tid = threadIdx.x;
  __shared__ float r1[256], r2[256], r3[256];
  float s1 = 0.f, s2 = 0.f, s3 = 0.f;
  for (int t = tid; t < 1024; t += 256) {
    s1 += flowP[b * 1024 + t];
    s2 += dualP[b * 1024 + t];
    s3 += gv[b * 1024 + t] * dem[b * 1024 + t];
  }
  r1[tid] = s1; r2[tid] = s2; r3[tid] = s3;
  __syncthreads();
  for (int st = 128; st >= 1; st >>= 1) {
    if (tid < st) { r1[tid] += r1[tid + st]; r2[tid] += r2[tid + st]; r3[tid] += r3[tid + st]; }
    __syncthreads();
  }
  if (tid == 0) out[b] = r1[0] - r2[0] + r3[0];
}

// ============================================================================
static inline char* carve(char*& p, size_t bytes) {
  char* r = p;
  p += (bytes + 255) & ~(size_t)255;
  return r;
}

extern "C" void kernel_launch(void* const* d_in, const int* in_sizes, int n_in,
                              void* d_out, int out_size, void* d_ws, size_t ws_size,
                              hipStream_t stream) {
  (void)in_sizes; (void)n_in; (void)out_size; (void)ws_size;
  const float* feat = (const float*)d_in[0];
  const float* emb  = (const float*)d_in[1];
  const float* dem  = (const float*)d_in[2];
  const float* adj  = (const float*)d_in[3];
  const float* nbr  = (const float*)d_in[4];
  const int*   cf   = (const int*)  d_in[5];
  // params: JAX tree-flatten (sorted-key) order
  const float* Uh   = (const float*)d_in[6];
  const float* Ur   = (const float*)d_in[7];
  const float* Uz   = (const float*)d_in[8];
  const float* Wh   = (const float*)d_in[9];
  const float* Wn   = (const float*)d_in[10];
  const float* Wo   = (const float*)d_in[11];
  const float* Wr   = (const float*)d_in[12];
  const float* Wz   = (const float*)d_in[13];
  const float* attW = (const float*)d_in[14];
  const float* attv = (const float*)d_in[15];
  const float* bh   = (const float*)d_in[16];
  const float* bn   = (const float*)d_in[17];
  const float* bo   = (const float*)d_in[18];
  const float* br   = (const float*)d_in[19];
  const float* bz   = (const float*)d_in[20];
  const float* dW0  = (const float*)d_in[21];
  const float* dW1  = (const float*)d_in[22];
  const float* db0  = (const float*)d_in[23];
  const float* db1  = (const float*)d_in[24];
  const float* eW0  = (const float*)d_in[25];
  const float* eW1  = (const float*)d_in[26];
  const float* eb0  = (const float*)d_in[27];
  const float* eb1  = (const float*)d_in[28];
  const float* gW0  = (const float*)d_in[29];
  const float* gW1  = (const float*)d_in[30];
  const float* gb0  = (const float*)d_in[31];
  const float* gb1  = (const float*)d_in[32];

  char* p = (char*)d_ws;
  float*    enc   = (float*)   carve(p, (size_t)NB * NN * ND * 4);      // 2 MB
  float*    nxt   = (float*)   carve(p, (size_t)NB * NN * ND * 4);      // 2 MB
  float*    agg   = (float*)   carve(p, (size_t)2 * NB * NN * ND * 4);  // 4 MB
  _Float16* tp    = (_Float16*)carve(p, (size_t)NB * NN * ND * 2);      // 1 MB
  float*    nw    = (float*)   carve(p, (size_t)NB * NN * 4);
  float*    gv    = (float*)   carve(p, (size_t)NB * NN * 4);
  float*    tauv  = (float*)   carve(p, (size_t)NB * NN * 4);
  float*    sA    = (float*)   carve(p, (size_t)NB * NN * 4);
  float*    sB    = (float*)   carve(p, (size_t)NB * NN * 4);
  float*    flowP = (float*)   carve(p, (size_t)NB * NN * 4);
  float*    dualP = (float*)   carve(p, (size_t)NB * NN * 4);

  const int NODES = NB * NN;  // 8192

  k_encoder<<<NODES, 64, 0, stream>>>(emb, feat, eW0, eb0, eW1, eb1, enc);

  for (int layer = 0; layer < 2; ++layer) {
    k_prep_t<<<NODES, 64, 0, stream>>>(enc, Wn, bn, tp);
    k_agg<<<dim3(16, NB, 2), 128, 0, stream>>>(nbr, tp, agg);
    k_attn<<<NODES, 64, 0, stream>>>(agg, attW, attv, Wo, bo, nxt);
    k_gru<<<NODES, 64, 0, stream>>>(nxt, enc, Wz, Uz, bz, Wr, Ur, br, Wh, Uh, bh);
  }

  k_heads<<<NODES, 64, 0, stream>>>(enc, dW0, db0, dW1, db1, gW0, gb0, gW1, gb1, nw, gv);
  k_sparsemax<<<NODES / 8, 256, 0, stream>>>(adj, nw, tauv);

  k_zero<<<(NODES + 255) / 256, 256, 0, stream>>>(sA, NODES);
  float* si = sA;
  float* so = sB;
  for (int it = 0; it < 10; ++it) {
    k_flow<<<dim3(32, NB), 256, 0, stream>>>(adj, nw, tauv, dem, si, so);
    float* t = si; si = so; so = t;
  }
  // final scale vector is in `si`

  k_flowcost<<<dim3(32, 32, NB), dim3(32, 8), 0, stream>>>(adj, nw, tauv, si, cf, flowP);
  k_dualcost<<<dim3(32, 32, NB), dim3(32, 8), 0, stream>>>(adj, gv, dualP);
  k_final<<<NB, 256, 0, stream>>>(flowP, dualP, gv, dem, (float*)d_out);
}
